// GCNEncoder_26336739459289
// MI455X (gfx1250) — compile-verified, matrix operand order checked by reading, not run on hardware
//
#include <hip/hip_runtime.h>
#include <stdint.h>

// ---------------------------------------------------------------------------
// Types for gfx1250 WMMA / async-LDS
// ---------------------------------------------------------------------------
typedef __attribute__((ext_vector_type(16))) __bf16 v16bf;
typedef __attribute__((ext_vector_type(8)))  __bf16 v8bf;
typedef __attribute__((ext_vector_type(8)))  float  v8f;
typedef int v4i_ __attribute__((vector_size(4 * sizeof(int))));

#if defined(__gfx1250__) && \
    __has_builtin(__builtin_amdgcn_global_load_async_to_lds_b128) && \
    __has_builtin(__builtin_amdgcn_s_wait_asynccnt)
#define HAVE_ASYNC_LDS 1
#else
#define HAVE_ASYNC_LDS 0
#endif

// ---------------------------------------------------------------------------
// bf16 helpers (bit-exact, no reliance on __bf16 arithmetic codegen)
// ---------------------------------------------------------------------------
__device__ __forceinline__ __bf16 bits_to_bf16(unsigned short b) {
    return __builtin_bit_cast(__bf16, b);
}
__device__ __forceinline__ float bf16_bits_to_f32(unsigned short b) {
    unsigned int u = ((unsigned int)b) << 16;
    return __builtin_bit_cast(float, u);
}
__device__ __forceinline__ unsigned short f32_to_bf16_rne_bits(float f) {
    unsigned int u = __builtin_bit_cast(unsigned int, f);
    unsigned int r = u + 0x7FFFu + ((u >> 16) & 1u); // round to nearest even
    return (unsigned short)(r >> 16);
}

// ---------------------------------------------------------------------------
// Small utility kernels
// ---------------------------------------------------------------------------
__global__ void k_fill1(float* __restrict__ p, int n) {
    int i = blockIdx.x * blockDim.x + threadIdx.x;
    if (i < n) p[i] = 1.0f;   // self-loop contributes degree 1 to every node
}

__global__ void k_degree(const long long* __restrict__ dst, float* __restrict__ deg, int E) {
    int e = blockIdx.x * blockDim.x + threadIdx.x;
    if (e < E) atomicAdd(&deg[(int)dst[e]], 1.0f);
}

__global__ void k_rsqrt(const float* __restrict__ deg, float* __restrict__ dinv, int n) {
    int i = blockIdx.x * blockDim.x + threadIdx.x;
    if (i < n) dinv[i] = rsqrtf(deg[i]);  // deg >= 1 always (self loops)
}

// split fp32 -> bf16 hi + bf16 lo  (x ~= hi + lo, ~16 mantissa bits)
__global__ void k_split_bf16(const float* __restrict__ in, __bf16* __restrict__ hi,
                             __bf16* __restrict__ lo, long long n) {
    long long i = (long long)blockIdx.x * blockDim.x + threadIdx.x;
    if (i >= n) return;
    float v = in[i];
    unsigned short hb = f32_to_bf16_rne_bits(v);
    float hv = bf16_bits_to_f32(hb);
    hi[i] = bits_to_bf16(hb);
    lo[i] = bits_to_bf16(f32_to_bf16_rne_bits(v - hv));
}

// transpose + split: in[K][N] (row-major) -> thi/tlo[N][K] (bf16 hi/lo)
// output index o = n*K + k  (coalesced writes); K is a power of two
__global__ void k_split_bf16_T(const float* __restrict__ in, __bf16* __restrict__ thi,
                               __bf16* __restrict__ tlo, int N, int log2K, long long total) {
    long long o = (long long)blockIdx.x * blockDim.x + threadIdx.x;
    if (o >= total) return;
    int n = (int)(o >> log2K);
    int k = (int)(o & ((1 << log2K) - 1));
    float v = in[(size_t)k * N + n];
    unsigned short hb = f32_to_bf16_rne_bits(v);
    thi[o] = bits_to_bf16(hb);
    tlo[o] = bits_to_bf16(f32_to_bf16_rne_bits(v - bf16_bits_to_f32(hb)));
}

// ---------------------------------------------------------------------------
// Split-bf16 (3x) WMMA GEMM with LDS-staged A panel:
//   C[M,N] = A[M,K] * B[K,N],  B given TRANSPOSED as Bt[N][K]
//   A ~ Ahi+Alo, B ~ Bhi+Blo;  C = AhiBhi + AhiBlo + AloBhi (f32 accumulate)
//
// One block per 16-row panel (tm); 8 waves stride over the 16-col tiles (tn).
// The contiguous 16xK hi/lo A panel (32KB, K=512) is staged in LDS once per
// block via GLOBAL_LOAD_ASYNC_TO_LDS_B128 (fallback: b128 load + ds_store),
// then every fragment is two aligned 16B chunks:
//   ISA 7.12.2 (16-bit A 16x32): lane L row m=L&15, half=L>>4,
//     elems j=0..7  -> K = k0 + 8*half + (0..7)
//     elems j=8..15 -> K = k0 + 16 + 8*half + (0..7)
//   B 32x16 mirrors the K pattern per column n=L&15 -> contiguous in Bt rows.
//   C/D f32 16x16: lane col n=L&15, vgpr j -> row j + 8*half.
// No divergent branches anywhere -> EXEC all-ones around WMMA (ISA 7.12 req).
// Requires K == 512 (host guarantees), M%16==0, (N/16)%8==0.
// ---------------------------------------------------------------------------
__global__ __launch_bounds__(256)
void k_gemm_lds_bf16x3(const __bf16* __restrict__ Ahi, const __bf16* __restrict__ Alo,
                       const __bf16* __restrict__ Bthi, const __bf16* __restrict__ Btlo,
                       float* __restrict__ C, int K, int N) {
    __shared__ __bf16 sAhi[16 * 512];
    __shared__ __bf16 sAlo[16 * 512];

    const int tm   = blockIdx.x;
    const int lane = threadIdx.x & 31;
    const int wave = threadIdx.x >> 5;
    const int half = lane >> 4;      // 0: lanes 0-15, 1: lanes 16-31
    const int m    = lane & 15;      // A row within tile / B,C column within tile
    const int tilesN = N >> 4;

    // ---- stage the contiguous 16-row A panel (hi+lo) into LDS ----
    const __bf16* gAhi = Ahi + (size_t)tm * 16 * K;
    const __bf16* gAlo = Alo + (size_t)tm * 16 * K;
    const int chunks = (16 * K) >> 3;              // #16-byte chunks per matrix
    for (int c = threadIdx.x; c < chunks; c += 256) {
#if HAVE_ASYNC_LDS
        __builtin_amdgcn_global_load_async_to_lds_b128(
            (v4i_*)(gAhi + c * 8), (v4i_*)(sAhi + c * 8), 0, 0);
        __builtin_amdgcn_global_load_async_to_lds_b128(
            (v4i_*)(gAlo + c * 8), (v4i_*)(sAlo + c * 8), 0, 0);
#else
        *(float4*)(sAhi + c * 8) = *(const float4*)(gAhi + c * 8);
        *(float4*)(sAlo + c * 8) = *(const float4*)(gAlo + c * 8);
#endif
    }
#if HAVE_ASYNC_LDS
    __builtin_amdgcn_s_wait_asynccnt(0);
#endif
    __syncthreads();

    const __bf16* lahi = sAhi + m * K;   // this lane's A row (LDS)
    const __bf16* lalo = sAlo + m * K;

    for (int tn = wave; tn < tilesN; tn += 8) {
        const __bf16* brow_hi = Bthi + (size_t)(tn * 16 + m) * K;  // Bt row = column of B
        const __bf16* brow_lo = Btlo + (size_t)(tn * 16 + m) * K;
        v8f acc = {};
        for (int k0 = 0; k0 < K; k0 += 32) {
            __builtin_prefetch(brow_hi + k0 + 128, 0, 0);   // global_prefetch_b8
            __builtin_prefetch(brow_lo + k0 + 128, 0, 0);
            const int o0 = k0 + 8 * half;
            const int o1 = k0 + 16 + 8 * half;
            v8bf a0 = *(const v8bf*)(lahi + o0);     // ds_load_b128
            v8bf a1 = *(const v8bf*)(lahi + o1);
            v8bf l0 = *(const v8bf*)(lalo + o0);
            v8bf l1 = *(const v8bf*)(lalo + o1);
            v8bf b0 = *(const v8bf*)(brow_hi + o0);  // global_load_b128
            v8bf b1 = *(const v8bf*)(brow_hi + o1);
            v8bf c0 = *(const v8bf*)(brow_lo + o0);
            v8bf c1 = *(const v8bf*)(brow_lo + o1);
            v16bf ahi = __builtin_shufflevector(a0, a1, 0,1,2,3,4,5,6,7,8,9,10,11,12,13,14,15);
            v16bf alo = __builtin_shufflevector(l0, l1, 0,1,2,3,4,5,6,7,8,9,10,11,12,13,14,15);
            v16bf bhi = __builtin_shufflevector(b0, b1, 0,1,2,3,4,5,6,7,8,9,10,11,12,13,14,15);
            v16bf blo = __builtin_shufflevector(c0, c1, 0,1,2,3,4,5,6,7,8,9,10,11,12,13,14,15);
            // (Ahi+Alo)(Bhi+Blo) ~= AhiBhi + AhiBlo + AloBhi  (drop lo*lo ~ 2^-16)
            acc = __builtin_amdgcn_wmma_f32_16x16x32_bf16(false, ahi, false, bhi,
                                                          (short)0, acc, false, false);
            acc = __builtin_amdgcn_wmma_f32_16x16x32_bf16(false, ahi, false, blo,
                                                          (short)0, acc, false, false);
            acc = __builtin_amdgcn_wmma_f32_16x16x32_bf16(false, alo, false, bhi,
                                                          (short)0, acc, false, false);
        }
        // store D: vgpr j -> row tm*16 + j + 8*half, col tn*16 + m
#pragma unroll
        for (int j = 0; j < 8; ++j) {
            int row = tm * 16 + j + 8 * half;
            C[(size_t)row * N + tn * 16 + m] = acc[j];
        }
    }
}

// ---------------------------------------------------------------------------
// Edge scatter:  acc[dst] += h[src] * dinv[src]*dinv[dst]
// thread -> (edge, 4-feature chunk); consecutive threads share an edge for
// coalesced float4 gathers and clustered atomics. chunks = F/4 is pow2.
// ---------------------------------------------------------------------------
__global__ __launch_bounds__(256)
void k_scatter(const float* __restrict__ h, const float* __restrict__ dinv,
               const long long* __restrict__ src, const long long* __restrict__ dst,
               float* __restrict__ acc, long long total, int log2chunks, int F) {
    long long gid = (long long)blockIdx.x * blockDim.x + threadIdx.x;
    if (gid >= total) return;
    int e  = (int)(gid >> log2chunks);
    int c4 = (int)(gid & ((1 << log2chunks) - 1)) << 2;
    int s = (int)src[e];
    int d = (int)dst[e];
    float norm = dinv[s] * dinv[d];
    const float4 hv = *(const float4*)(h + (size_t)s * F + c4);
    float* ap = acc + (size_t)d * F + c4;
    atomicAdd(ap + 0, hv.x * norm);
    atomicAdd(ap + 1, hv.y * norm);
    atomicAdd(ap + 2, hv.z * norm);
    atomicAdd(ap + 3, hv.w * norm);
}

// layer-1 finalize: relu(acc + selfloop + bias), re-split to bf16 hi/lo
__global__ void k_finalize_relu_split(const float* __restrict__ acc, const float* __restrict__ hpre,
                                      const float* __restrict__ dinv, const float* __restrict__ bias,
                                      __bf16* __restrict__ hi, __bf16* __restrict__ lo,
                                      long long total, int log2F) {
    long long i = (long long)blockIdx.x * blockDim.x + threadIdx.x;
    if (i >= total) return;
    int n = (int)(i >> log2F);
    int c = (int)(i & ((1 << log2F) - 1));
    float di = dinv[n];
    float v = acc[i] + hpre[i] * di * di + bias[c];
    v = fmaxf(v, 0.0f);
    unsigned short hb = f32_to_bf16_rne_bits(v);
    hi[i] = bits_to_bf16(hb);
    lo[i] = bits_to_bf16(f32_to_bf16_rne_bits(v - bf16_bits_to_f32(hb)));
}

// layer-2 finalize: acc + selfloop + bias -> d_out (fp32)
__global__ void k_finalize_out(const float* __restrict__ acc, const float* __restrict__ hpre,
                               const float* __restrict__ dinv, const float* __restrict__ bias,
                               float* __restrict__ out, long long total, int log2F) {
    long long i = (long long)blockIdx.x * blockDim.x + threadIdx.x;
    if (i >= total) return;
    int n = (int)(i >> log2F);
    int c = (int)(i & ((1 << log2F) - 1));
    float di = dinv[n];
    out[i] = acc[i] + hpre[i] * di * di + bias[c];
}

// ---------------------------------------------------------------------------
// Host launcher
// ---------------------------------------------------------------------------
static inline unsigned int blocks_for(long long total, int bs) {
    return (unsigned int)((total + bs - 1) / bs);
}

extern "C" void kernel_launch(void* const* d_in, const int* in_sizes, int n_in,
                              void* d_out, int out_size, void* d_ws, size_t ws_size,
                              hipStream_t stream) {
    const float*     x  = (const float*)d_in[0];          // [N, 512]
    const long long* ei = (const long long*)d_in[1];      // [2, E] int64
    const float*     W1 = (const float*)d_in[2];          // [512, 512]
    const float*     b1 = (const float*)d_in[3];          // [512]
    const float*     W2 = (const float*)d_in[4];          // [512, 256]
    const float*     b2 = (const float*)d_in[5];          // [256]

    const int IN  = 512;
    const int HID = in_sizes[2] / IN;            // 512
    const int OUT = in_sizes[4] / HID;           // 256
    const int Nn  = in_sizes[0] / IN;            // 50000
    const int E   = in_sizes[1] / 2;             // 800000
    const long long* src = ei;
    const long long* dst = ei + E;

    // ---------------- workspace carve-out ----------------
    char* base = (char*)d_ws;
    size_t off = 0;
    auto carve = [&](size_t bytes) -> void* {
        off = (off + 255) & ~(size_t)255;
        void* p = base + off;
        off += bytes;
        return p;
    };
    float*  deg    = (float*)carve((size_t)Nn * 4);
    float*  dinv   = (float*)carve((size_t)Nn * 4);
    __bf16* w1thi  = (__bf16*)carve((size_t)IN * HID * 2);   // W1^T [HID][IN]
    __bf16* w1tlo  = (__bf16*)carve((size_t)IN * HID * 2);
    __bf16* w2thi  = (__bf16*)carve((size_t)HID * OUT * 2);  // W2^T [OUT][HID]
    __bf16* w2tlo  = (__bf16*)carve((size_t)HID * OUT * 2);
    __bf16* xhi    = (__bf16*)carve((size_t)Nn * IN * 2);    // reused as h1hi
    __bf16* xlo    = (__bf16*)carve((size_t)Nn * IN * 2);    // reused as h1lo
    float*  hpre1  = (float*)carve((size_t)Nn * HID * 4);    // reused as hpre2
    float*  acc1   = (float*)carve((size_t)Nn * HID * 4);    // reused as acc2
    __bf16* h1hi   = xhi;
    __bf16* h1lo   = xlo;
    float*  hpre2  = hpre1;
    float*  acc2   = acc1;

    const int BS = 256;

    // ---------------- degree / normalization ----------------
    k_fill1<<<blocks_for(Nn, BS), BS, 0, stream>>>(deg, Nn);
    k_degree<<<blocks_for(E, BS), BS, 0, stream>>>(dst, deg, E);
    k_rsqrt<<<blocks_for(Nn, BS), BS, 0, stream>>>(deg, dinv, Nn);

    // ---------------- bf16 hi/lo splits (weights transposed to N-major) ----
    k_split_bf16_T<<<blocks_for((long long)IN * HID, BS), BS, 0, stream>>>(
        W1, w1thi, w1tlo, HID, 9, (long long)IN * HID);      // K=IN=512
    k_split_bf16_T<<<blocks_for((long long)HID * OUT, BS), BS, 0, stream>>>(
        W2, w2thi, w2tlo, OUT, 9, (long long)HID * OUT);     // K=HID=512
    k_split_bf16<<<blocks_for((long long)Nn * IN, BS), BS, 0, stream>>>(
        x, xhi, xlo, (long long)Nn * IN);

    (void)hipMemsetAsync(acc1, 0, (size_t)Nn * HID * 4, stream);

    // ---------------- layer 1 ----------------
    // GEMM1: one block per 16-row panel; waves stride the 32 column tiles
    k_gemm_lds_bf16x3<<<Nn / 16, 256, 0, stream>>>(xhi, xlo, w1thi, w1tlo, hpre1, IN, HID);
    {   // edge scatter, F=512 -> 128 chunks (log2=7)
        long long total = (long long)E * (HID / 4);
        k_scatter<<<blocks_for(total, BS), BS, 0, stream>>>(hpre1, dinv, src, dst, acc1, total, 7, HID);
    }
    {   // relu + bias + self-loop, split for layer 2 (log2F=9)
        long long total = (long long)Nn * HID;
        k_finalize_relu_split<<<blocks_for(total, BS), BS, 0, stream>>>(acc1, hpre1, dinv, b1,
                                                                       h1hi, h1lo, total, 9);
    }

    (void)hipMemsetAsync(acc2, 0, (size_t)Nn * OUT * 4, stream);

    // ---------------- layer 2 ----------------
    k_gemm_lds_bf16x3<<<Nn / 16, 256, 0, stream>>>(h1hi, h1lo, w2thi, w2tlo, hpre2, HID, OUT);
    {   // edge scatter, F=256 -> 64 chunks (log2=6)
        long long total = (long long)E * (OUT / 4);
        k_scatter<<<blocks_for(total, BS), BS, 0, stream>>>(hpre2, dinv, src, dst, acc2, total, 6, OUT);
    }
    {   // final bias + self-loop -> d_out (log2F=8)
        long long total = (long long)Nn * OUT;
        k_finalize_out<<<blocks_for(total, BS), BS, 0, stream>>>(acc2, hpre2, dinv, b2,
                                                                (float*)d_out, total, 8);
    }
}
